// MultiTaskHead_26268019982741
// MI455X (gfx1250) — compile-verified, matrix operand order checked by reading, not run on hardware
//
#include <hip/hip_runtime.h>
#include <hip/hip_bf16.h>
#include <math.h>

typedef _Float16 half_t;
typedef __attribute__((ext_vector_type(16))) _Float16 v16h;
typedef __attribute__((ext_vector_type(8)))  _Float16 v8h;
typedef __attribute__((ext_vector_type(8)))  float    v8f;

#define DEV static __device__ __forceinline__

constexpr int D_MODEL = 1024;
constexpr int N_DEV   = 64;
constexpr int HIDDEN  = 128;
constexpr int DK      = 128;   // head dim = 1024/8
constexpr int BB      = 16;    // batch
constexpr int SS      = 1024;  // sequence

// ---------------- WMMA fragment helpers (CDNA5 16x16x32 f16) ----------------
// A 16x32 (f16): lane m=lane&15 holds row m; kg=lane>>4.
// Element j<8 -> K = kg*8 + j ; j>=8 -> K = 16 + kg*8 + (j-8).
// => per lane: two contiguous 8-half (16B) runs at +kg*8 and +16+kg*8.
DEV v16h load_afrag(const half_t* base, int ld) {
  int lane = threadIdx.x & 31;
  int m = lane & 15, kg = lane >> 4;
  const half_t* p = base + (size_t)m * ld + kg * 8;
  v8h lo = *(const v8h*)p;
  v8h hi = *(const v8h*)(p + 16);
  return __builtin_shufflevector(lo, hi, 0, 1, 2, 3, 4, 5, 6, 7,
                                 8, 9, 10, 11, 12, 13, 14, 15);
}

DEV v8f wmma16(v16h a, v16h b, v8f c) {
  return __builtin_amdgcn_wmma_f32_16x16x32_f16(
      /*neg_a=*/false, a, /*neg_b=*/false, b,
      /*c_mod=*/(short)0, c, /*reuse_a=*/false, /*reuse_b=*/false);
}

// ---------------- conversions / padding helpers ----------------
__global__ void f32_to_f16(const float* __restrict__ src, half_t* __restrict__ dst,
                           size_t n) {
  size_t i = (size_t)blockIdx.x * blockDim.x + threadIdx.x;
  if (i < n) dst[i] = (half_t)src[i];
}

__global__ void zero_f16(half_t* __restrict__ dst, size_t n) {
  size_t i = (size_t)blockIdx.x * blockDim.x + threadIdx.x;
  if (i < n) dst[i] = (half_t)0.f;
}

// src [rows, ci] f32 -> dst [rows, co] f16 with zero-padded extra columns
__global__ void f32_to_f16_padc(const float* __restrict__ src, half_t* __restrict__ dst,
                                int rows, int ci, int co) {
  size_t total = (size_t)rows * co;
  size_t idx = (size_t)blockIdx.x * blockDim.x + threadIdx.x;
  if (idx >= total) return;
  int r = (int)(idx / co), c = (int)(idx % co);
  dst[idx] = (c < ci) ? (half_t)src[(size_t)r * ci + c] : (half_t)0.f;
}

// transpose-convert stacked weights: src [n_mats, rows, cols] f32 ->
// dst [n_mats, cols, rows] f16  (write-coalesced)
__global__ void f32_to_f16_T(const float* __restrict__ src, half_t* __restrict__ dst,
                             int n_mats, int rows, int cols) {
  size_t total = (size_t)n_mats * rows * cols;
  size_t idx = (size_t)blockIdx.x * blockDim.x + threadIdx.x;
  if (idx >= total) return;
  int k = (int)(idx % rows);
  int c = (int)((idx / rows) % cols);
  int n = (int)(idx / ((size_t)rows * cols));
  dst[idx] = (half_t)src[((size_t)n * rows + k) * cols + c];
}

// V [B*S, D] (f16) -> Vt [B, H, DK, S] (f16): Vt[((b*8+h)*128+d)*1024 + s]
__global__ void transpose_v(const half_t* __restrict__ V, half_t* __restrict__ Vt) {
  size_t idx = (size_t)blockIdx.x * blockDim.x + threadIdx.x;
  if (idx >= (size_t)BB * SS * D_MODEL) return;
  int s = (int)(idx & 1023);
  int d = (int)((idx >> 10) & 127);
  int h = (int)((idx >> 17) & 7);
  int b = (int)(idx >> 20);
  Vt[idx] = V[((size_t)b * SS + s) * D_MODEL + h * DK + d];
}

// ---------------- tiled WMMA GEMM: C = A(f16) @ B(f16) [+ bias] ----------------
// REQUIRES: M % 128 == 0 (grid.y = M/128), N % 128 == 0 (grid.x = N/128), K % 32 == 0.
// 128x128 block tile, BK=32, 256 threads = 8 waves (4x2), wave tile 32x64.
// B tile staged TRANSPOSED in LDS so both A and B fragments are 2x b128 loads.
// Templated on bias/output type so the epilogue is branch-free.
template <bool HAS_BIAS, bool F16OUT>
__global__ __launch_bounds__(256)
void gemm_f16_wmma(const half_t* __restrict__ A, int lda,
                   const half_t* __restrict__ Bm, int ldb,
                   const float* __restrict__ bias,
                   void* __restrict__ Cout, int ldc, int K) {
  __shared__ __align__(16) half_t As[128 * 40];   // [row][k], ld=40
  __shared__ __align__(16) half_t Bst[128 * 40];  // [col][k], ld=40 (B transposed)
  int tid = threadIdx.x, lane = tid & 31, wid = tid >> 5;
  int wm = wid & 3, wn = wid >> 2;
  int bm0 = blockIdx.y * 128, bn0 = blockIdx.x * 128;
  v8f acc[2][4];
#pragma unroll
  for (int i = 0; i < 2; ++i)
#pragma unroll
    for (int j = 0; j < 4; ++j)
#pragma unroll
      for (int e = 0; e < 8; ++e) acc[i][j][e] = 0.f;

  for (int k0 = 0; k0 < K; k0 += 32) {
    // A tile: 128x32 halves = 512 x v8h; 2 vectors per thread
#pragma unroll
    for (int i = 0; i < 2; ++i) {
      int vec = i * 256 + tid;
      int r = vec >> 2, c = (vec & 3) * 8;
      v8h av = *(const v8h*)&A[(size_t)(bm0 + r) * lda + k0 + c];
      *(v8h*)&As[r * 40 + c] = av;
    }
    // B tile: 32x128, stored transposed: Bst[col][krow]
#pragma unroll
    for (int i = 0; i < 2; ++i) {
      int vec = i * 256 + tid;
      int r = vec >> 4, c = (vec & 15) * 8;  // r: k-row 0..31, c: col 0..120
      v8h bv = *(const v8h*)&Bm[(size_t)(k0 + r) * ldb + bn0 + c];
#pragma unroll
      for (int e = 0; e < 8; ++e) Bst[(c + e) * 40 + r] = bv[e];
    }
    __syncthreads();
    v16h af[2], bf[4];
#pragma unroll
    for (int i = 0; i < 2; ++i) af[i] = load_afrag(&As[(wm * 32 + i * 16) * 40], 40);
#pragma unroll
    for (int j = 0; j < 4; ++j) bf[j] = load_afrag(&Bst[(wn * 64 + j * 16) * 40], 40);
#pragma unroll
    for (int i = 0; i < 2; ++i)
#pragma unroll
      for (int j = 0; j < 4; ++j) acc[i][j] = wmma16(af[i], bf[j], acc[i][j]);
    __syncthreads();
  }

  int mb = (lane >> 4) * 8, nn = lane & 15;
#pragma unroll
  for (int i = 0; i < 2; ++i)
#pragma unroll
    for (int j = 0; j < 4; ++j) {
      int gm0 = bm0 + wm * 32 + i * 16, gn0 = bn0 + wn * 64 + j * 16;
      int col = gn0 + nn;
      float bv = HAS_BIAS ? bias[col] : 0.f;
#pragma unroll
      for (int r = 0; r < 8; ++r) {
        int row = gm0 + mb + r;
        float v = acc[i][j][r] + bv;
        if (F16OUT)
          ((half_t*)Cout)[(size_t)row * ldc + col] = (half_t)v;
        else
          ((float*)Cout)[(size_t)row * ldc + col] = v;
      }
    }
}

// ---------------- flash-style attention, one wave per (b, h, 16-query tile) ----------------
// Q,K: [B*S, D] f16 (head at col h*DK); Vt: [B,H,DK,S] f16 (pre-transposed)
__global__ __launch_bounds__(32)
void attn_wmma(const half_t* __restrict__ Q, const half_t* __restrict__ K,
               const half_t* __restrict__ Vt, half_t* __restrict__ ctx) {
  __shared__ __align__(16) half_t Ps[16 * 40];
  int qt = blockIdx.x, h = blockIdx.y, b = blockIdx.z;
  int lane = threadIdx.x & 31;
  int q0 = qt * 16;
  const float scale = 0.08838834764831845f;  // 1/sqrt(128)

  const half_t* Qp = Q + ((size_t)b * SS + q0) * D_MODEL + h * DK;
  v16h qf[4];
#pragma unroll
  for (int kk = 0; kk < 4; ++kk) qf[kk] = load_afrag(Qp + kk * 32, D_MODEL);

  v8f ct[8];
#pragma unroll
  for (int t = 0; t < 8; ++t)
#pragma unroll
    for (int r = 0; r < 8; ++r) ct[t][r] = 0.f;
  float rm[8], rl[8];
#pragma unroll
  for (int r = 0; r < 8; ++r) { rm[r] = -1e30f; rl[r] = 0.f; }

  const half_t* Vbase = Vt + (size_t)(b * 8 + h) * DK * SS;

  for (int kb = 0; kb < SS; kb += 32) {
    v8f s[2];
#pragma unroll
    for (int t = 0; t < 2; ++t) {
#pragma unroll
      for (int r = 0; r < 8; ++r) s[t][r] = 0.f;
      // K^T as B-operand == A-fragment gather of K rows (key per lane)
      const half_t* Kp = K + ((size_t)b * SS + kb + t * 16) * D_MODEL + h * DK;
#pragma unroll
      for (int kk = 0; kk < 4; ++kk) {
        v16h kf = load_afrag(Kp + kk * 32, D_MODEL);
        s[t] = wmma16(qf[kk], kf, s[t]);
      }
    }
    // online softmax over the 32-key block; C layout: row = r+8*(lane>>4), N = lane&15
#pragma unroll
    for (int r = 0; r < 8; ++r) {
      float a0 = s[0][r] * scale, a1 = s[1][r] * scale;
      float tm = fmaxf(a0, a1);
      tm = fmaxf(tm, __shfl_xor(tm, 1, 32));
      tm = fmaxf(tm, __shfl_xor(tm, 2, 32));
      tm = fmaxf(tm, __shfl_xor(tm, 4, 32));
      tm = fmaxf(tm, __shfl_xor(tm, 8, 32));
      float newm = fmaxf(rm[r], tm);
      float fac = __expf(rm[r] - newm);
      float p0 = __expf(a0 - newm), p1 = __expf(a1 - newm);
      float rs = p0 + p1;
      rs += __shfl_xor(rs, 1, 32);
      rs += __shfl_xor(rs, 2, 32);
      rs += __shfl_xor(rs, 4, 32);
      rs += __shfl_xor(rs, 8, 32);
      rl[r] = rl[r] * fac + rs;
      rm[r] = newm;
#pragma unroll
      for (int t = 0; t < 8; ++t) ct[t][r] *= fac;
      int row = r + ((lane >> 4) << 3);
      int col = lane & 15;
      Ps[row * 40 + col]      = (half_t)p0;
      Ps[row * 40 + col + 16] = (half_t)p1;
    }
    __syncthreads();                   // single-wave WG: cheap
    v16h pf = load_afrag(&Ps[0], 40);  // P 16x32 as A-fragment
#pragma unroll
    for (int t = 0; t < 8; ++t) {
      // V^T tile [d, key] -> A-fragment layout (lane = d, contiguous keys)
      v16h vf = load_afrag(Vbase + (size_t)(t * 16) * SS + kb, SS);
      ct[t] = wmma16(pf, vf, ct[t]);
    }
    __syncthreads();
  }

#pragma unroll
  for (int t = 0; t < 8; ++t)
#pragma unroll
    for (int r = 0; r < 8; ++r) {
      int row = q0 + r + ((lane >> 4) << 3);
      int col = h * DK + t * 16 + (lane & 15);
      ctx[((size_t)b * SS + row) * D_MODEL + col] = (half_t)(ct[t][r] / rl[r]);
    }
}

// ---------------- mean over sequence: g = mean_s pooled ----------------
__global__ void mean_pool(const half_t* __restrict__ pooled, float* __restrict__ g,
                          half_t* __restrict__ gh) {
  int idx = blockIdx.x * blockDim.x + threadIdx.x;  // b*D + d
  if (idx >= BB * D_MODEL) return;
  int b = idx / D_MODEL, d = idx % D_MODEL;
  const half_t* p = pooled + (size_t)b * SS * D_MODEL + d;
  float sum = 0.f;
  for (int s = 0; s < SS; ++s) sum += (float)p[(size_t)s * D_MODEL];
  sum *= (1.0f / SS);
  g[idx] = sum;
  gh[idx] = (half_t)sum;
}

// ---------------- h = tanh(G1[b] + E1[n] + b1) ----------------
__global__ void tanh_combine(const float* __restrict__ G1, const float* __restrict__ E1,
                             const float* __restrict__ b1, half_t* __restrict__ hout) {
  int idx = blockIdx.x * blockDim.x + threadIdx.x;
  if (idx >= BB * N_DEV * D_MODEL) return;
  int d = idx & (D_MODEL - 1);
  int bn = idx >> 10;
  int b = bn >> 6, n = bn & 63;
  hout[idx] = (half_t)tanhf(G1[b * D_MODEL + d] + E1[n * D_MODEL + d] + b1[d]);
}

// ---------------- row softmax over 64 logits (logits padded to ld=128) ----------------
__global__ void softmax64(const float* __restrict__ logits, const float* __restrict__ bias,
                          half_t* __restrict__ rw) {
  int row = blockIdx.x * blockDim.x + threadIdx.x;
  if (row >= BB * N_DEV) return;
  const float* l = logits + (size_t)row * 128;
  float m = -1e30f;
  for (int j = 0; j < N_DEV; ++j) m = fmaxf(m, l[j] + bias[j]);
  float s = 0.f;
  for (int j = 0; j < N_DEV; ++j) s += __expf(l[j] + bias[j] - m);
  float inv = 1.f / s;
  for (int j = 0; j < N_DEV; ++j)
    rw[(size_t)row * N_DEV + j] = (half_t)(__expf(l[j] + bias[j] - m) * inv);
}

// ---------------- combined = g_exp + device_repr (f16 out) ----------------
__global__ void add_combine(const float* __restrict__ g, const float* __restrict__ devrep,
                            half_t* __restrict__ comb) {
  int idx = blockIdx.x * blockDim.x + threadIdx.x;
  if (idx >= BB * N_DEV * D_MODEL) return;
  int d = idx & (D_MODEL - 1);
  int b = (idx >> 10) >> 6;
  comb[idx] = (half_t)(g[b * D_MODEL + d] + devrep[idx]);
}

// ---------------- per-device head: relu(C@W1+b1)·w2 + b2 -> softplus/sigmoid ----------------
// w1t: [N_DEV, HIDDEN, D] f16 (pre-transposed) so B-fragments are contiguous b128 loads.
__global__ __launch_bounds__(32)
void head_wmma(const half_t* __restrict__ comb,  // [B*N_DEV, D] rows b*64+n
               const half_t* __restrict__ w1t,   // [N_DEV, HIDDEN, D] f16
               const float* __restrict__ b1,     // [N_DEV, HIDDEN]
               const float* __restrict__ w2,     // [N_DEV, HIDDEN]
               const float* __restrict__ b2,     // [N_DEV]
               float* __restrict__ out,          // [B, N_DEV]
               int is_cls) {
  int n = blockIdx.x;
  int lane = threadIdx.x & 31;
  const half_t* A = comb + (size_t)n * D_MODEL;  // row b at stride N_DEV*D
  const half_t* W = w1t + (size_t)n * HIDDEN * D_MODEL;
  v8f acc[8];
#pragma unroll
  for (int t = 0; t < 8; ++t)
#pragma unroll
    for (int r = 0; r < 8; ++r) acc[t][r] = 0.f;

  for (int k0 = 0; k0 < D_MODEL; k0 += 32) {
    v16h af = load_afrag(A + k0, N_DEV * D_MODEL);
#pragma unroll
    for (int t = 0; t < 8; ++t) {
      v16h bf = load_afrag(W + (size_t)(t * 16) * D_MODEL + k0, D_MODEL);
      acc[t] = wmma16(af, bf, acc[t]);
    }
  }

  int nn = lane & 15, mb = (lane >> 4) * 8;
  float part[8];
#pragma unroll
  for (int r = 0; r < 8; ++r) {
    float p = 0.f;
#pragma unroll
    for (int t = 0; t < 8; ++t) {
      float hv = acc[t][r] + b1[n * HIDDEN + t * 16 + nn];
      hv = fmaxf(hv, 0.f);
      p += hv * w2[n * HIDDEN + t * 16 + nn];
    }
    p += __shfl_xor(p, 1, 32);
    p += __shfl_xor(p, 2, 32);
    p += __shfl_xor(p, 4, 32);
    p += __shfl_xor(p, 8, 32);
    part[r] = p;
  }
  if (nn == 0) {
#pragma unroll
    for (int r = 0; r < 8; ++r) {
      int b = mb + r;
      float v = part[r] + b2[n];
      v = is_cls ? (1.f / (1.f + __expf(-v)))
                 : (v > 20.f ? v : log1pf(__expf(v)));
      out[b * N_DEV + n] = v;
    }
  }
}

// ---------------- host-side orchestration ----------------
extern "C" void kernel_launch(void* const* d_in, const int* in_sizes, int n_in,
                              void* d_out, int out_size, void* d_ws, size_t ws_size,
                              hipStream_t stream) {
  const float* x      = (const float*)d_in[0];
  const float* wq     = (const float*)d_in[1];
  const float* bq     = (const float*)d_in[2];
  const float* wk     = (const float*)d_in[3];
  const float* bk     = (const float*)d_in[4];
  const float* wv     = (const float*)d_in[5];
  const float* bv     = (const float*)d_in[6];
  const float* wo     = (const float*)d_in[7];
  const float* bo     = (const float*)d_in[8];
  const float* emb    = (const float*)d_in[9];
  const float* rg_w1  = (const float*)d_in[10];
  const float* rg_b1  = (const float*)d_in[11];
  const float* rg_w2  = (const float*)d_in[12];
  const float* rg_b2  = (const float*)d_in[13];
  const float* reg_w1 = (const float*)d_in[14];
  const float* reg_b1 = (const float*)d_in[15];
  const float* reg_w2 = (const float*)d_in[16];
  const float* reg_b2 = (const float*)d_in[17];
  const float* cls_w1 = (const float*)d_in[18];
  const float* cls_b1 = (const float*)d_in[19];
  const float* cls_w2 = (const float*)d_in[20];
  const float* cls_b2 = (const float*)d_in[21];
  float* out_reg = (float*)d_out;
  float* out_cls = out_reg + BB * N_DEV;

  char* ws = (char*)d_ws;
  size_t off = 0;
  auto carve = [&](size_t bytes) -> void* {
    void* p = ws + off;
    off = (off + bytes + 255) & ~(size_t)255;
    return p;
  };
  const size_t NX = (size_t)BB * SS * D_MODEL;  // 16.7M
  half_t* xh      = (half_t*)carve(NX * 2);
  half_t* wqh     = (half_t*)carve((size_t)D_MODEL * D_MODEL * 2);
  half_t* wkh     = (half_t*)carve((size_t)D_MODEL * D_MODEL * 2);
  half_t* wvh     = (half_t*)carve((size_t)D_MODEL * D_MODEL * 2);
  half_t* woh     = (half_t*)carve((size_t)D_MODEL * D_MODEL * 2);
  half_t* Qh      = (half_t*)carve(NX * 2);     // reused as pooled (f16) later
  half_t* Kh      = (half_t*)carve(NX * 2);
  half_t* Vh      = (half_t*)carve(NX * 2);
  half_t* Vth     = (half_t*)carve(NX * 2);     // V transposed [B,H,DK,S]
  half_t* ctxh    = (half_t*)carve(NX * 2);
  float*  g       = (float*)carve((size_t)BB * D_MODEL * 4);
  half_t* ghp     = (half_t*)carve((size_t)128 * D_MODEL * 2);      // g f16 padded to 128 rows
  half_t* embp    = (half_t*)carve((size_t)128 * D_MODEL * 2);      // emb f16 padded to 128 rows
  half_t* rgw1h   = (half_t*)carve((size_t)2 * D_MODEL * D_MODEL * 2);
  half_t* rgw2p   = (half_t*)carve((size_t)D_MODEL * 128 * 2);      // rg_w2 padded to 128 cols
  half_t* regw1t  = (half_t*)carve((size_t)N_DEV * D_MODEL * HIDDEN * 2);
  half_t* clsw1t  = (half_t*)carve((size_t)N_DEV * D_MODEL * HIDDEN * 2);
  float*  G1p     = (float*)carve((size_t)128 * D_MODEL * 4);       // padded rows
  float*  E1p     = (float*)carve((size_t)128 * D_MODEL * 4);       // padded rows
  half_t* hth     = (half_t*)carve((size_t)BB * N_DEV * D_MODEL * 2);
  float*  logitsp = (float*)carve((size_t)BB * N_DEV * 128 * 4);    // padded cols
  half_t* rwh     = (half_t*)carve((size_t)BB * N_DEV * N_DEV * 2);
  float*  devrep  = (float*)carve((size_t)BB * N_DEV * D_MODEL * 4);
  half_t* combh   = (half_t*)carve((size_t)BB * N_DEV * D_MODEL * 2);

  auto cvt = [&](const float* s, half_t* d, size_t n) {
    f32_to_f16<<<dim3((unsigned)((n + 255) / 256)), dim3(256), 0, stream>>>(s, d, n);
  };
  cvt(x, xh, NX);
  cvt(wq, wqh, (size_t)D_MODEL * D_MODEL);
  cvt(wk, wkh, (size_t)D_MODEL * D_MODEL);
  cvt(wv, wvh, (size_t)D_MODEL * D_MODEL);
  cvt(wo, woh, (size_t)D_MODEL * D_MODEL);
  cvt(rg_w1, rgw1h, (size_t)2 * D_MODEL * D_MODEL);
  // emb -> padded 128-row f16 (rows 64..127 zero, required every call)
  cvt(emb, embp, (size_t)N_DEV * D_MODEL);
  zero_f16<<<dim3((unsigned)((64 * D_MODEL + 255) / 256)), dim3(256), 0, stream>>>(
      embp + (size_t)N_DEV * D_MODEL, (size_t)64 * D_MODEL);
  // gh padded rows 16..127 zero
  zero_f16<<<dim3((unsigned)((112 * D_MODEL + 255) / 256)), dim3(256), 0, stream>>>(
      ghp + (size_t)BB * D_MODEL, (size_t)112 * D_MODEL);
  // rg_w2 [1024,64] -> [1024,128] zero-padded cols
  f32_to_f16_padc<<<dim3((unsigned)((D_MODEL * 128 + 255) / 256)), dim3(256), 0, stream>>>(
      rg_w2, rgw2p, D_MODEL, N_DEV, 128);
  {
    size_t n = (size_t)N_DEV * D_MODEL * HIDDEN;
    dim3 gsz((unsigned)((n + 255) / 256));
    f32_to_f16_T<<<gsz, dim3(256), 0, stream>>>(reg_w1, regw1t, N_DEV, D_MODEL, HIDDEN);
    f32_to_f16_T<<<gsz, dim3(256), 0, stream>>>(cls_w1, clsw1t, N_DEV, D_MODEL, HIDDEN);
  }

  // GEMM variants: bias + f16 out (projections), no-bias + f32 out (routing gate)
  auto gemm_bias_f16 = [&](const half_t* A, int lda, const half_t* Bm, int ldb,
                           const float* bias, half_t* C, int ldc, int M, int N, int K) {
    dim3 grid((unsigned)(N / 128), (unsigned)(M / 128));
    gemm_f16_wmma<true, true><<<grid, dim3(256), 0, stream>>>(A, lda, Bm, ldb, bias,
                                                              (void*)C, ldc, K);
  };
  auto gemm_f32 = [&](const half_t* A, int lda, const half_t* Bm, int ldb,
                      float* C, int ldc, int M, int N, int K) {
    dim3 grid((unsigned)(N / 128), (unsigned)(M / 128));
    gemm_f16_wmma<false, false><<<grid, dim3(256), 0, stream>>>(A, lda, Bm, ldb, nullptr,
                                                                (void*)C, ldc, K);
  };

  const int MX = BB * SS;  // 16384
  // QKV projections (f16 outputs for attention)
  gemm_bias_f16(xh, D_MODEL, wqh, D_MODEL, bq, Qh, D_MODEL, MX, D_MODEL, D_MODEL);
  gemm_bias_f16(xh, D_MODEL, wkh, D_MODEL, bk, Kh, D_MODEL, MX, D_MODEL, D_MODEL);
  gemm_bias_f16(xh, D_MODEL, wvh, D_MODEL, bv, Vh, D_MODEL, MX, D_MODEL, D_MODEL);

  // transpose V for contiguous PV B-fragments
  transpose_v<<<dim3((unsigned)((NX + 255) / 256)), dim3(256), 0, stream>>>(Vh, Vth);

  // attention -> ctx (f16)
  attn_wmma<<<dim3(SS / 16, 8, BB), dim3(32), 0, stream>>>(Qh, Kh, Vth, ctxh);

  // pooled = ctx @ wo + bo (f16, overwrite Qh which is dead now)
  half_t* pooledh = Qh;
  gemm_bias_f16(ctxh, D_MODEL, woh, D_MODEL, bo, pooledh, D_MODEL, MX, D_MODEL, D_MODEL);

  // g = mean over sequence (writes rows 0..15 of ghp; tail already zeroed)
  mean_pool<<<dim3((BB * D_MODEL + 255) / 256), dim3(256), 0, stream>>>(pooledh, g, ghp);

  // routing gate: split ri@rg_w1 into g@W1a + emb@W1b (padded M=128)
  gemm_f32(ghp, D_MODEL, rgw1h, D_MODEL, G1p, D_MODEL, 128, D_MODEL, D_MODEL);
  gemm_f32(embp, D_MODEL, rgw1h + (size_t)D_MODEL * D_MODEL, D_MODEL, E1p, D_MODEL,
           128, D_MODEL, D_MODEL);
  tanh_combine<<<dim3((BB * N_DEV * D_MODEL + 255) / 256), dim3(256), 0, stream>>>(
      G1p, E1p, rg_b1, hth);

  // logits = h @ rg_w2 (padded N=128, bias folded into softmax); softmax; rw @ emb
  gemm_f32(hth, D_MODEL, rgw2p, 128, logitsp, 128, BB * N_DEV, 128, D_MODEL);
  softmax64<<<dim3((BB * N_DEV + 255) / 256), dim3(256), 0, stream>>>(logitsp, rg_b2, rwh);
  gemm_f32(rwh, N_DEV, embp, D_MODEL, devrep, D_MODEL, BB * N_DEV, D_MODEL, N_DEV);
  add_combine<<<dim3((BB * N_DEV * D_MODEL + 255) / 256), dim3(256), 0, stream>>>(
      g, devrep, combh);

  // per-device heads
  head_wmma<<<dim3(N_DEV), dim3(32), 0, stream>>>(combh, regw1t, reg_b1, reg_w2,
                                                  reg_b2, out_reg, 0);
  head_wmma<<<dim3(N_DEV), dim3(32), 0, stream>>>(combh, clsw1t, cls_b1, cls_w2,
                                                  cls_b2, out_cls, 1);
  (void)in_sizes; (void)n_in; (void)out_size; (void)ws_size;
}
